// BracketMultitaskNet_16535624090421
// MI455X (gfx1250) — matrix operand order repaired and checked by reading, not compiled, not run on hardware
//
#include <hip/hip_runtime.h>

// ---------------- types ----------------
typedef __bf16 bf16_t;
typedef bf16_t v16bf __attribute__((ext_vector_type(16)));
typedef bf16_t v8bf  __attribute__((ext_vector_type(8)));
typedef float  v8f   __attribute__((ext_vector_type(8)));

#define S_DIM 256
#define B_DIM 2048
#define D_DIM 128
#define D2_DIM 256

// Branch-free GELU (tanh form), built on native v_exp_f32 (exp2) + v_rcp_f32.
// gelu(x) = 0.5*x*(1 + tanh(0.79788456*(x + 0.044715*x^3)))
// tanh(u) = sign(u) * (1 - 2/(exp(2|u|)+1));  exp(2|u|) = exp2(|u| * 2/ln2)
static __device__ __forceinline__ float gelu_(float x) {
    float u  = 0.7978845608028654f * x * __builtin_fmaf(0.044715f * x, x, 1.0f);
    float au = __builtin_fabsf(u);
    float e  = __builtin_amdgcn_exp2f(au * 2.8853900817779268f);   // exp(2|u|)
    float th = __builtin_fmaf(-2.0f, __builtin_amdgcn_rcpf(e + 1.0f), 1.0f);
    th = __builtin_copysignf(th, u);
    return 0.5f * x * (1.0f + th);
}

static __device__ __forceinline__ v8f wmma_bf16(v16bf a, v16bf b, v8f c) {
    return __builtin_amdgcn_wmma_f32_16x16x32_bf16(
        /*neg_a=*/false, a, /*neg_b=*/false, b,
        /*c_mod=*/(short)0, c, /*reuse_a=*/false, /*reuse_b=*/false);
}

// load 8 consecutive fp32 and convert to bf16 (RNE)
static __device__ __forceinline__ v8bf cvt8(const float* p) {
    const float4 a = *(const float4*)p;
    const float4 b = *(const float4*)(p + 4);
    v8bf r;
    r[0] = (bf16_t)a.x; r[1] = (bf16_t)a.y; r[2] = (bf16_t)a.z; r[3] = (bf16_t)a.w;
    r[4] = (bf16_t)b.x; r[5] = (bf16_t)b.y; r[6] = (bf16_t)b.z; r[7] = (bf16_t)b.w;
    return r;
}

static __device__ __forceinline__ v16bf cat16(v8bf lo, v8bf hi) {
    return __builtin_shufflevector(lo, hi, 0,1,2,3,4,5,6,7,8,9,10,11,12,13,14,15);
}

// ---------------- weight prep: fp32 -> bf16, transposed to N-major ----------------
// ws layout (bf16 elements):
//   [0      .. 32768) : W1t [N=256][K=128]   (from W1 [128][256])
//   [32768  .. 65536) : W2t [N=128][K=256]   (from W2 [256][128])
//   [65536  .. 98304) : Wbt [N=128][K=256]   (from Wb [256][128])
__global__ void prep_weights(const float* __restrict__ W1,
                             const float* __restrict__ W2,
                             const float* __restrict__ Wb,
                             bf16_t* __restrict__ ws) {
    int i = blockIdx.x * blockDim.x + threadIdx.x;
    if (i >= 128 * 256) return;
    int k1 = i >> 8, n1 = i & 255;                 // W1 [k][n]
    ws[n1 * 128 + k1] = (bf16_t)W1[i];
    int k2 = i >> 7, n2 = i & 127;                 // W2/Wb [k][n]
    ws[32768 + n2 * 256 + k2] = (bf16_t)W2[i];
    ws[65536 + n2 * 256 + k2] = (bf16_t)Wb[i];
}

// ---------------- fused main kernel ----------------
// block = 256 threads = 8 waves; wave w handles 16 rows (fixed s, b-tile)
// grid = (B/128, S)
__global__ __launch_bounds__(256)
void fused_bracket_net(const float* __restrict__ src,
                       const float* __restrict__ bb,
                       const float* __restrict__ b1,
                       const float* __restrict__ b2,
                       const float* __restrict__ gamma,
                       const float* __restrict__ beta,
                       const bf16_t* __restrict__ W1t,
                       const bf16_t* __restrict__ W2t,
                       const bf16_t* __restrict__ Wbt,
                       float* __restrict__ out) {
    __shared__ __align__(16) bf16_t Hbf[8 * 16 * D2_DIM];   // 64 KB: gelu(X@W1+b1) per wave

    const int wave = threadIdx.x >> 5;
    const int lane = threadIdx.x & 31;
    const int half = lane >> 4;        // 0: lanes 0-15, 1: lanes 16-31
    const int nl   = lane & 15;
    const int s    = blockIdx.y;
    const int brow = blockIdx.x * 128 + wave * 16;
    const size_t g0 = (size_t)s * B_DIM + (size_t)brow;     // first global row of tile

    float* __restrict__ out_ret   = out;
    float* __restrict__ out_diffs = out + (size_t)S_DIM * B_DIM * D_DIM;
    float* __restrict__ out_cond  = out_diffs + (size_t)(S_DIM - 1) * B_DIM * D_DIM;

    if (blockIdx.x == 0 && blockIdx.y == 0 && threadIdx.x < 3)
        out_cond[threadIdx.x] = 0.0f;   // rate=0 branches never fire

    // ---- A fragments of src[s] tile (16x128 bf16), contiguous per-lane k order (k = 16*half + s) ----
    v16bf Atop[4];
    {
        const float* arow = src + (g0 + (size_t)nl) * D_DIM;
        #pragma unroll
        for (int kk = 0; kk < 4; ++kk) {
            v8bf lo = cvt8(arow + kk * 32 + 16 * half);
            v8bf hi = cvt8(arow + kk * 32 + 16 * half + 8);
            Atop[kk] = cat16(lo, hi);
        }
    }

    bf16_t* Hw = Hbf + wave * (16 * D2_DIM);

    // ================= GEMM1: H = gelu(X @ W1 + b1), 16x256 ==================
    #pragma unroll 2
    for (int t = 0; t < 16; ++t) {
        v8f acc = {};
        const bf16_t* wp = W1t + (size_t)(t * 16 + nl) * 128 + half * 16;
        #pragma unroll
        for (int kk = 0; kk < 4; ++kk)
            acc = wmma_bf16(Atop[kk], *(const v16bf*)(wp + kk * 32), acc);
        const float bias = b1[t * 16 + nl];
        #pragma unroll
        for (int v = 0; v < 8; ++v) {
            float h = gelu_(acc[v] + bias);
            Hw[(v + 8 * half) * D2_DIM + t * 16 + nl] = (bf16_t)h;
        }
    }

    // ================= GEMM3: diffs = gelu([src[s],src[s+1]] @ Wb + bb) ======
    if (s < S_DIM - 1) {
        v16bf Abot[4];
        const float* arow2 = src + (g0 + B_DIM + (size_t)nl) * D_DIM;
        #pragma unroll
        for (int kk = 0; kk < 4; ++kk) {
            v8bf lo = cvt8(arow2 + kk * 32 + 16 * half);
            v8bf hi = cvt8(arow2 + kk * 32 + 16 * half + 8);
            Abot[kk] = cat16(lo, hi);
        }
        #pragma unroll 2
        for (int t = 0; t < 8; ++t) {
            v8f acc = {};
            const bf16_t* wp = Wbt + (size_t)(t * 16 + nl) * 256 + half * 16;
            #pragma unroll
            for (int kk = 0; kk < 4; ++kk)
                acc = wmma_bf16(Atop[kk], *(const v16bf*)(wp + kk * 32), acc);
            #pragma unroll
            for (int kk = 0; kk < 4; ++kk)
                acc = wmma_bf16(Abot[kk], *(const v16bf*)(wp + 128 + kk * 32), acc);
            const float bias = bb[t * 16 + nl];
            #pragma unroll
            for (int v = 0; v < 8; ++v)
                out_diffs[(g0 + v + 8 * half) * (size_t)D_DIM + t * 16 + nl] =
                    gelu_(acc[v] + bias);
        }
    }

    // ================= GEMM2: tf = H @ W2 + b2; r = gelu(tf)+src; LayerNorm ==
    v16bf Ah[8];
    {
        const bf16_t* hp = Hw + nl * D2_DIM;
        #pragma unroll
        for (int kk = 0; kk < 8; ++kk) {
            v8bf lo = *(const v8bf*)(hp + kk * 32 + 16 * half);
            v8bf hi = *(const v8bf*)(hp + kk * 32 + 16 * half + 8);
            Ah[kk] = cat16(lo, hi);
        }
    }

    v8f r[8];
    #pragma unroll 2
    for (int t = 0; t < 8; ++t) {
        v8f acc = {};
        const bf16_t* wp = W2t + (size_t)(t * 16 + nl) * 256 + half * 16;
        #pragma unroll
        for (int kk = 0; kk < 8; ++kk)
            acc = wmma_bf16(Ah[kk], *(const v16bf*)(wp + kk * 32), acc);
        const float bias = b2[t * 16 + nl];
        #pragma unroll
        for (int v = 0; v < 8; ++v) {
            float sv = src[(g0 + v + 8 * half) * (size_t)D_DIM + t * 16 + nl];
            acc[v] = gelu_(acc[v] + bias) + sv;
        }
        r[t] = acc;
    }

    // LayerNorm over D=128: row m = v + 8*half lives in one 16-lane group.
    float mu[8], rstd[8];
    #pragma unroll
    for (int v = 0; v < 8; ++v) {
        float sm = 0.0f, sq = 0.0f;
        #pragma unroll
        for (int t = 0; t < 8; ++t) { float x = r[t][v]; sm += x; sq += x * x; }
        #pragma unroll
        for (int m = 1; m < 16; m <<= 1) {
            sm += __shfl_xor(sm, m, 32);
            sq += __shfl_xor(sq, m, 32);
        }
        float mean = sm * 0.0078125f;                 // /128
        float var  = sq * 0.0078125f - mean * mean;
        mu[v]   = mean;
        rstd[v] = rsqrtf(var + 1e-5f);
    }
    #pragma unroll
    for (int t = 0; t < 8; ++t) {
        const float ga = gamma[t * 16 + nl];
        const float be = beta[t * 16 + nl];
        #pragma unroll
        for (int v = 0; v < 8; ++v)
            out_ret[(g0 + v + 8 * half) * (size_t)D_DIM + t * 16 + nl] =
                (r[t][v] - mu[v]) * rstd[v] * ga + be;
    }
}

// ---------------- launch ----------------
extern "C" void kernel_launch(void* const* d_in, const int* in_sizes, int n_in,
                              void* d_out, int out_size, void* d_ws, size_t ws_size,
                              hipStream_t stream) {
    (void)in_sizes; (void)n_in; (void)out_size; (void)ws_size;
    const float* src   = (const float*)d_in[0];
    const float* Wb    = (const float*)d_in[1];
    const float* bb    = (const float*)d_in[2];
    const float* W1    = (const float*)d_in[3];
    const float* b1    = (const float*)d_in[4];
    const float* W2    = (const float*)d_in[5];
    const float* b2    = (const float*)d_in[6];
    const float* gamma = (const float*)d_in[7];
    const float* beta  = (const float*)d_in[8];

    bf16_t* wbuf = (bf16_t*)d_ws;
    prep_weights<<<128, 256, 0, stream>>>(W1, W2, Wb, wbuf);

    dim3 grid(B_DIM / 128, S_DIM);
    fused_bracket_net<<<grid, 256, 0, stream>>>(
        src, bb, b1, b2, gamma, beta,
        wbuf,            // W1t [256][128]
        wbuf + 32768,    // W2t [128][256]
        wbuf + 65536,    // Wbt [128][256]
        (float*)d_out);
}
